// get_model_91276644974601
// MI455X (gfx1250) — matrix operand order repaired and checked by reading, not compiled, not run on hardware
//
#include <hip/hip_runtime.h>
#include <stdint.h>
#include <stddef.h>

// ---------------------------------------------------------------------------
// Types for WMMA (CDNA5 / gfx1250, wave32)
// ---------------------------------------------------------------------------
typedef __attribute__((ext_vector_type(16))) __bf16 v16bf;
typedef __attribute__((ext_vector_type(8)))  float  v8f;

#define BATCH 8
#define NPTS  8192

__host__ __device__ inline int pad32i(int x) { return (x + 31) & ~31; }

__device__ __forceinline__ uint16_t f2bf(float f) {
  union { float f; uint32_t u; } x; x.f = f;
  uint32_t r = x.u + 0x7FFFu + ((x.u >> 16) & 1u);   // round-to-nearest-even
  return (uint16_t)(r >> 16);
}
__device__ __forceinline__ float bf2f(uint16_t h) {
  union { uint32_t u; float f; } x; x.u = ((uint32_t)h) << 16; return x.f;
}

// ---------------------------------------------------------------------------
// 1. split xyz (B,6,N) -> coords (B,N,3), pts (B,N,3)
// ---------------------------------------------------------------------------
__global__ void split_xyz_k(const float* __restrict__ xyz, float* __restrict__ coords,
                            float* __restrict__ pts) {
  size_t t = (size_t)blockIdx.x * blockDim.x + threadIdx.x;
  size_t tot = (size_t)BATCH * NPTS * 3;
  if (t >= tot) return;
  int d = (int)(t % 3);
  int j = (int)((t / 3) % NPTS);
  int b = (int)(t / ((size_t)3 * NPTS));
  coords[t] = xyz[((size_t)b * 6 + d) * NPTS + j];
  pts[t]    = xyz[((size_t)b * 6 + 3 + d) * NPTS + j];
}

// ---------------------------------------------------------------------------
// 2. Farthest point sampling: one block per batch, dist array in LDS
// ---------------------------------------------------------------------------
__global__ void fps_k(const float* __restrict__ coords, int n, int npoint,
                      int* __restrict__ idx) {
  __shared__ float sdist[8192];
  __shared__ float rv[256];
  __shared__ int   ri[256];
  __shared__ int   sfar;
  int b = blockIdx.x, tid = threadIdx.x;
  const float* P = coords + (size_t)b * n * 3;
  for (int j = tid; j < n; j += 256) sdist[j] = 1e10f;
  if (tid == 0) sfar = 0;
  __syncthreads();
  for (int i = 0; i < npoint; ++i) {
    int far = sfar;
    if (tid == 0) idx[b * npoint + i] = far;
    float cx = P[far * 3 + 0], cy = P[far * 3 + 1], cz = P[far * 3 + 2];
    float bv = -1.0f; int bi = 0;
    for (int j = tid; j < n; j += 256) {
      float dx = P[j * 3 + 0] - cx, dy = P[j * 3 + 1] - cy, dz = P[j * 3 + 2] - cz;
      float d = dx * dx + dy * dy + dz * dz;
      float nd = fminf(sdist[j], d);
      sdist[j] = nd;
      if (nd > bv) { bv = nd; bi = j; }
    }
    rv[tid] = bv; ri[tid] = bi;
    __syncthreads();
    for (int o = 128; o > 0; o >>= 1) {
      if (tid < o) {
        if (rv[tid + o] > rv[tid] ||
            (rv[tid + o] == rv[tid] && ri[tid + o] < ri[tid])) {
          rv[tid] = rv[tid + o]; ri[tid] = ri[tid + o];
        }
      }
      __syncthreads();
    }
    if (tid == 0) sfar = ri[0];
    __syncthreads();
  }
}

// ---------------------------------------------------------------------------
// 3. gather centers: new_xyz[b,s,:] = coords[b, idx[b,s], :]
// ---------------------------------------------------------------------------
__global__ void gather_coords_k(const float* __restrict__ coords, const int* __restrict__ idx,
                                int n, int S, float* __restrict__ out) {
  size_t t = (size_t)blockIdx.x * blockDim.x + threadIdx.x;
  size_t tot = (size_t)BATCH * S * 3;
  if (t >= tot) return;
  int d = (int)(t % 3);
  int s = (int)((t / 3) % S);
  int b = (int)(t / ((size_t)3 * S));
  int gi = idx[b * S + s];
  out[t] = coords[((size_t)b * n + gi) * 3 + d];
}

// ---------------------------------------------------------------------------
// 4. ball query: one wave per (b,s); ballot-compaction preserves ascending idx
// ---------------------------------------------------------------------------
__global__ void query_ball_k(const float* __restrict__ coords, const float* __restrict__ nxyz,
                             int n, int S, int K, float r2, int* __restrict__ gidx) {
  int wid  = (int)(((size_t)blockIdx.x * blockDim.x + threadIdx.x) >> 5);
  int lane = threadIdx.x & 31;
  if (wid >= BATCH * S) return;
  int b = wid / S, s = wid % S;
  float cx = nxyz[((size_t)b * S + s) * 3 + 0];
  float cy = nxyz[((size_t)b * S + s) * 3 + 1];
  float cz = nxyz[((size_t)b * S + s) * 3 + 2];
  int* out = gidx + ((size_t)b * S + s) * K;
  const float* P = coords + (size_t)b * n * 3;
  int cnt = 0, first = -1;
  for (int j0 = 0; j0 < n && cnt < K; j0 += 32) {
    int j = j0 + lane;
    bool within = false;
    if (j < n) {
      float dx = P[j * 3 + 0] - cx, dy = P[j * 3 + 1] - cy, dz = P[j * 3 + 2] - cz;
      within = (dx * dx + dy * dy + dz * dz) <= r2;
    }
    unsigned mask = (unsigned)__ballot(within);
    if (first < 0 && mask) first = j0 + __builtin_ctz(mask);
    if (within) {
      int pos = cnt + __popc(mask & ((1u << lane) - 1u));
      if (pos < K) out[pos] = j;
    }
    cnt += __popc(mask);
  }
  if (cnt > K) cnt = K;
  if (first < 0) first = 0;
  for (int k = cnt + lane; k < K; k += 32) out[k] = first;
}

// ---------------------------------------------------------------------------
// 5. grouped feature build (per batch): feat[s*K+k, c] bf16, zero K-padded
//    feat = concat(points[gidx], coords[gidx]-new_xyz)
// ---------------------------------------------------------------------------
__global__ void group_feat_k(const float* __restrict__ coords, const float* __restrict__ nxyz,
                             const float* __restrict__ points, const int* __restrict__ gidx,
                             int b, int n, int S, int K, int inC, int cinP,
                             uint16_t* __restrict__ feat) {
  size_t t = (size_t)blockIdx.x * blockDim.x + threadIdx.x;
  size_t tot = (size_t)S * K * cinP;
  if (t >= tot) return;
  int c = (int)(t % cinP);
  size_t sk = t / cinP;
  int k = (int)(sk % K);
  int s = (int)(sk / K);
  int gi = gidx[((size_t)b * S + s) * K + k];
  if (gi >= n) gi = 0;
  float val = 0.0f;
  if (c < inC) {
    val = points[((size_t)b * n + gi) * inC + c];
  } else if (c < inC + 3) {
    int d = c - inC;
    val = coords[((size_t)b * n + gi) * 3 + d] - nxyz[((size_t)b * S + s) * 3 + d];
  }
  feat[t] = f2bf(val);
}

// ---------------------------------------------------------------------------
// 6. conv+BN prep: Wp bf16 padded [coutP, cinP]; scale/shift folded BN
// ---------------------------------------------------------------------------
__global__ void prep_conv_k(const float* __restrict__ W, const float* __restrict__ bias,
                            const float* __restrict__ beta, const float* __restrict__ g,
                            const float* __restrict__ mean, const float* __restrict__ var,
                            int cin, int cinP, int cout, int coutP,
                            uint16_t* __restrict__ Wp, float* __restrict__ scale,
                            float* __restrict__ shift) {
  size_t t = (size_t)blockIdx.x * blockDim.x + threadIdx.x;
  size_t tot = (size_t)coutP * cinP;
  if (t >= tot) return;
  int ci = (int)(t % cinP);
  int co = (int)(t / cinP);
  float w = (co < cout && ci < cin) ? W[(size_t)co * cin + ci] : 0.0f;
  Wp[t] = f2bf(w);
  if (ci == 0) {
    if (co < cout) {
      float sc = g[co] * rsqrtf(var[co] + 1e-5f);
      scale[co] = sc;
      shift[co] = (bias[co] - mean[co]) * sc + beta[co];
    } else {
      scale[co] = 0.0f; shift[co] = 0.0f;
    }
  }
}

// ---------------------------------------------------------------------------
// 7. WMMA GEMM + fused BN (+ReLU), N-blocked x4:
//    each wave computes a 16x64 output strip: one A fragment is reused
//    against 4 B fragments -> 4x v_wmma_f32_16x16x32_bf16 per K-step.
//    Out-of-range N tiles are clamped (redundant compute, not written).
// ---------------------------------------------------------------------------
__device__ __forceinline__ void gemm_epilogue(v8f acc, int nt, int nTiles, int mt,
                                              int lane, int halfSel,
                                              const float* __restrict__ scale,
                                              const float* __restrict__ shift,
                                              int cout, int coutP, int relu,
                                              uint16_t* __restrict__ outBf,
                                              float* __restrict__ outF32) {
  if (nt >= nTiles) return;                 // uniform per wave
  int ncol = (nt << 4) + (lane & 15);
  float sc = (ncol < cout) ? scale[ncol] : 0.0f;
  float sh = (ncol < cout) ? shift[ncol] : 0.0f;
#pragma unroll
  for (int q = 0; q < 8; ++q) {
    int m = (mt << 4) + q + (halfSel << 3);
    float val = acc[q] * sc + sh;
    if (relu) val = fmaxf(val, 0.0f);
    if (ncol >= cout) val = 0.0f;           // keep padded columns zero for next layer
    if (outBf)  outBf[(size_t)m * coutP + ncol] = f2bf(val);
    if (outF32 && ncol < cout) outF32[(size_t)m * cout + ncol] = val;
  }
}

__global__ void gemm_bn_k(const uint16_t* __restrict__ A, const uint16_t* __restrict__ Wp,
                          const float* __restrict__ scale, const float* __restrict__ shift,
                          int M, int cinP, int cout, int coutP,
                          uint16_t* __restrict__ outBf, float* __restrict__ outF32, int relu) {
  int wid  = (int)(((size_t)blockIdx.x * blockDim.x + threadIdx.x) >> 5);
  int lane = threadIdx.x & 31;
  int mTiles = M >> 4;
  int nTiles = coutP >> 4;
  int nGroups = (nTiles + 3) >> 2;
  if (wid >= mTiles * nGroups) return;      // uniform per wave: EXEC stays all-ones
  int mt  = wid % mTiles;
  int nt0 = (wid / mTiles) << 2;
  int halfSel = lane >> 4;                  // 0: lanes 0-15, 1: lanes 16-31

  // A fragment (16x32 bf16): lane holds row mt*16+(lane&15); K = k0+16*halfSel+s (contiguous)
  const uint16_t* arow = A + (size_t)((mt << 4) + (lane & 15)) * cinP + halfSel * 16;

  // 4 B fragment base pointers (32x16 bf16 each); clamp tile index to stay in
  // bounds -> branch-free inner loop, redundant tiles discarded in epilogue.
  const uint16_t* bp[4];
#pragma unroll
  for (int t = 0; t < 4; ++t) {
    int nt = nt0 + t; if (nt >= nTiles) nt = nTiles - 1;
    bp[t] = Wp + (size_t)((nt << 4) + (lane & 15)) * cinP + halfSel * 16;
  }
  const uint16_t* b0p = bp[0];
  const uint16_t* b1p = bp[1];
  const uint16_t* b2p = bp[2];
  const uint16_t* b3p = bp[3];

  v8f acc0 = {0.f, 0.f, 0.f, 0.f, 0.f, 0.f, 0.f, 0.f};
  v8f acc1 = acc0, acc2 = acc0, acc3 = acc0;

  for (int k0 = 0; k0 < cinP; k0 += 32) {
    union { uint32_t u[8]; v16bf v; } af, b0, b1, b2, b3;
#pragma unroll
    for (int q = 0; q < 4; ++q) {
      af.u[q]     = *(const uint32_t*)(arow + k0 + 2 * q);        // K = base+2q
      af.u[q + 4] = *(const uint32_t*)(arow + k0 + 2 * q + 8);    // K = base+8+2q
    }
#pragma unroll
    for (int q = 0; q < 8; ++q) {
      b0.u[q] = *(const uint32_t*)(b0p + k0 + 2 * q);             // K = kb+2q
      b1.u[q] = *(const uint32_t*)(b1p + k0 + 2 * q);
      b2.u[q] = *(const uint32_t*)(b2p + k0 + 2 * q);
      b3.u[q] = *(const uint32_t*)(b3p + k0 + 2 * q);
    }
    acc0 = __builtin_amdgcn_wmma_f32_16x16x32_bf16(false, af.v, false, b0.v, (short)0, acc0, false, false);
    acc1 = __builtin_amdgcn_wmma_f32_16x16x32_bf16(false, af.v, false, b1.v, (short)0, acc1, false, false);
    acc2 = __builtin_amdgcn_wmma_f32_16x16x32_bf16(false, af.v, false, b2.v, (short)0, acc2, false, false);
    acc3 = __builtin_amdgcn_wmma_f32_16x16x32_bf16(false, af.v, false, b3.v, (short)0, acc3, false, false);
  }

  gemm_epilogue(acc0, nt0 + 0, nTiles, mt, lane, halfSel, scale, shift, cout, coutP, relu, outBf, outF32);
  gemm_epilogue(acc1, nt0 + 1, nTiles, mt, lane, halfSel, scale, shift, cout, coutP, relu, outBf, outF32);
  gemm_epilogue(acc2, nt0 + 2, nTiles, mt, lane, halfSel, scale, shift, cout, coutP, relu, outBf, outF32);
  gemm_epilogue(acc3, nt0 + 3, nTiles, mt, lane, halfSel, scale, shift, cout, coutP, relu, outBf, outF32);
}

// ---------------------------------------------------------------------------
// 8. max over K samples -> pooled[b, s, cOff+c]
// ---------------------------------------------------------------------------
__global__ void maxpool_k(const uint16_t* __restrict__ X, int S, int K, int cout, int coutP,
                          float* __restrict__ pooled, int b, int Ctot, int cOff) {
  size_t t = (size_t)blockIdx.x * blockDim.x + threadIdx.x;
  size_t tot = (size_t)S * cout;
  if (t >= tot) return;
  int c = (int)(t % cout);
  int s = (int)(t / cout);
  float m = -3.4e38f;
  for (int k = 0; k < K; ++k)
    m = fmaxf(m, bf2f(X[((size_t)s * K + k) * coutP + c]));
  pooled[((size_t)b * S + s) * Ctot + cOff + c] = m;
}

// ---------------------------------------------------------------------------
// 9. f32 -> padded bf16 matrix
// ---------------------------------------------------------------------------
__global__ void f32_to_bf16_k(const float* __restrict__ in, int M, int C, int CP,
                              uint16_t* __restrict__ out) {
  size_t t = (size_t)blockIdx.x * blockDim.x + threadIdx.x;
  size_t tot = (size_t)M * CP;
  if (t >= tot) return;
  int c = (int)(t % CP);
  size_t m = t / CP;
  out[t] = f2bf(c < C ? in[m * C + c] : 0.0f);
}

// ---------------------------------------------------------------------------
// 10. softmax over S (points axis) per (b, channel)
// ---------------------------------------------------------------------------
__global__ void softmax_s_k(const float* __restrict__ x, float* __restrict__ w, int S, int C) {
  int bk = blockIdx.x;
  int b = bk / C, k = bk % C, tid = threadIdx.x;
  __shared__ float red[128];
  const float* xp = x + (size_t)b * S * C + k;
  float m = -3.4e38f;
  for (int s = tid; s < S; s += 128) m = fmaxf(m, xp[(size_t)s * C]);
  red[tid] = m; __syncthreads();
  for (int o = 64; o > 0; o >>= 1) {
    if (tid < o) red[tid] = fmaxf(red[tid], red[tid + o]);
    __syncthreads();
  }
  m = red[0]; __syncthreads();
  float sum = 0.0f;
  for (int s = tid; s < S; s += 128) sum += expf(xp[(size_t)s * C] - m);
  red[tid] = sum; __syncthreads();
  for (int o = 64; o > 0; o >>= 1) {
    if (tid < o) red[tid] += red[tid + o];
    __syncthreads();
  }
  float inv = 1.0f / red[0];
  float* wp = w + (size_t)b * S * C + k;
  for (int s = tid; s < S; s += 128) wp[(size_t)s * C] = expf(xp[(size_t)s * C] - m) * inv;
}

// ---------------------------------------------------------------------------
// 11. skeleton einsum: skel[b,k,d] = sum_s w[b,s,k] * coords[b,s,d]
// ---------------------------------------------------------------------------
__global__ void skel_out_k(const float* __restrict__ w, const float* __restrict__ coords,
                           float* __restrict__ out, int S, int C) {
  size_t t = (size_t)blockIdx.x * blockDim.x + threadIdx.x;
  size_t tot = (size_t)BATCH * C * 3;
  if (t >= tot) return;
  int d = (int)(t % 3);
  int k = (int)((t / 3) % C);
  int b = (int)(t / ((size_t)3 * C));
  float acc = 0.0f;
  for (int s = 0; s < S; ++s)
    acc += w[((size_t)b * S + s) * C + k] * coords[((size_t)b * S + s) * 3 + d];
  out[t] = acc;
}

// ---------------------------------------------------------------------------
// 12. weight transpose: wT[b,k,s] = w[b,s,k]
// ---------------------------------------------------------------------------
__global__ void w_transpose_k(const float* __restrict__ w, float* __restrict__ out, int S, int C) {
  size_t t = (size_t)blockIdx.x * blockDim.x + threadIdx.x;
  size_t tot = (size_t)BATCH * C * S;
  if (t >= tot) return;
  int s = (int)(t % S);
  int k = (int)((t / S) % C);
  int b = (int)(t / ((size_t)S * C));
  out[t] = w[((size_t)b * S + s) * C + k];
}

// ---------------------------------------------------------------------------
// Host orchestration
// ---------------------------------------------------------------------------
extern "C" void kernel_launch(void* const* d_in, const int* in_sizes, int n_in,
                              void* d_out, int out_size, void* d_ws, size_t ws_size,
                              hipStream_t stream) {
  (void)in_sizes; (void)n_in; (void)out_size; (void)ws_size;
  const float* xyz = (const float*)d_in[0];

  static const int   NP[4]      = {1024, 768, 512, 512};
  static const int   INCH[4]    = {3, 64, 128, 256};
  static const float RAD[4][2]  = {{0.1f, 0.2f}, {0.2f, 0.4f}, {0.4f, 0.6f}, {0.6f, 0.8f}};
  static const int   NS[4][2]   = {{16, 32}, {32, 64}, {32, 64}, {64, 128}};
  static const int   MLP[4][3]  = {{16, 16, 32}, {32, 32, 64}, {64, 64, 128}, {128, 128, 256}};
  static const int   HEAD[5][2] = {{512, 384}, {384, 256}, {256, 256}, {256, 128}, {128, 100}};

  // conv table: 24 SA convs (layer-major, scale, conv) + 5 head convs
  int convCin[29], convCout[29];
  {
    int j = 0;
    for (int l = 0; l < 4; ++l)
      for (int s = 0; s < 2; ++s) {
        int ci = INCH[l] + 3;
        for (int c = 0; c < 3; ++c) { convCin[j] = ci; convCout[j] = MLP[l][c]; ci = MLP[l][c]; ++j; }
      }
    for (int i = 0; i < 5; ++i) { convCin[j] = HEAD[i][0]; convCout[j] = HEAD[i][1]; ++j; }
  }

  // ---- workspace bump allocator ----
  char* wsb = (char*)d_ws;
  size_t off = 0;
  auto alloc = [&](size_t bytes) -> char* {
    off = (off + 255) & ~(size_t)255;
    char* p = wsb + off;
    off += bytes;
    return p;
  };

  int npl[5] = {NPTS, 1024, 768, 512, 512};
  int pcl[5] = {3, 64, 128, 256, 512};
  float* coordsL[5];
  coordsL[0] = (float*)alloc((size_t)BATCH * NPTS * 3 * 4);
  for (int l = 1; l < 5; ++l) coordsL[l] = (float*)alloc((size_t)BATCH * npl[l] * 3 * 4);
  float* pooled[5];
  pooled[0] = (float*)alloc((size_t)BATCH * NPTS * 3 * 4);   // pts0
  for (int l = 1; l < 5; ++l) pooled[l] = (float*)alloc((size_t)BATCH * npl[l] * pcl[l] * 4);
  int* fpsIdx = (int*)alloc((size_t)BATCH * 1024 * 4);
  int* gidx   = (int*)alloc((size_t)BATCH * 512 * 128 * 4);  // covers max B*S*K

  uint16_t* WpA[29]; float* scA[29]; float* shA[29];
  for (int j = 0; j < 29; ++j) {
    int cinP = pad32i(convCin[j]), coutP = pad32i(convCout[j]);
    WpA[j] = (uint16_t*)alloc((size_t)coutP * cinP * 2);
    scA[j] = (float*)alloc((size_t)coutP * 4);
    shA[j] = (float*)alloc((size_t)coutP * 4);
  }
  float* logits = (float*)alloc((size_t)BATCH * 512 * 100 * 4);
  float* wbuf   = (float*)alloc((size_t)BATCH * 512 * 100 * 4);
  uint16_t* FEAT = (uint16_t*)alloc((size_t)512 * 128 * 288 * 2);  // max S*K*cinP
  uint16_t* BUF1 = (uint16_t*)alloc((size_t)512 * 128 * 256 * 2);  // max S*K*coutP
  uint16_t* BUF2 = (uint16_t*)alloc((size_t)512 * 128 * 256 * 2);

  auto blocks = [](size_t tot) { return (unsigned)((tot + 255) / 256); };

  // ---- fold BN + pad weights (param leaves in pytree order: W,b,beta,g,m,v) ----
  for (int j = 0; j < 29; ++j) {
    const float* W  = (const float*)d_in[2 + j * 6 + 0];
    const float* bb = (const float*)d_in[2 + j * 6 + 1];
    const float* be = (const float*)d_in[2 + j * 6 + 2];
    const float* g  = (const float*)d_in[2 + j * 6 + 3];
    const float* mm = (const float*)d_in[2 + j * 6 + 4];
    const float* vv = (const float*)d_in[2 + j * 6 + 5];
    int cinP = pad32i(convCin[j]), coutP = pad32i(convCout[j]);
    size_t tot = (size_t)coutP * cinP;
    prep_conv_k<<<blocks(tot), 256, 0, stream>>>(W, bb, be, g, mm, vv,
        convCin[j], cinP, convCout[j], coutP, WpA[j], scA[j], shA[j]);
  }

  auto gemmLaunch = [&](const uint16_t* A, int j, int M, uint16_t* outBf,
                        float* outF32, int relu) {
    int cinP = pad32i(convCin[j]);
    int cout = convCout[j];
    int coutP = pad32i(cout);
    int mTiles = M / 16;
    int nGroups = (coutP / 16 + 3) / 4;
    int waves = mTiles * nGroups;
    gemm_bn_k<<<(waves + 7) / 8, 256, 0, stream>>>(A, WpA[j], scA[j], shA[j],
        M, cinP, cout, coutP, outBf, outF32, relu);
  };

  // ---- split input ----
  split_xyz_k<<<blocks((size_t)BATCH * NPTS * 3), 256, 0, stream>>>(xyz, coordsL[0], pooled[0]);

  // ---- set-abstraction layers ----
  const float* curC = coordsL[0];
  const float* curP = pooled[0];
  int curN = NPTS;
  for (int l = 0; l < 4; ++l) {
    int S = NP[l];
    fps_k<<<BATCH, 256, 0, stream>>>(curC, curN, S, fpsIdx);
    gather_coords_k<<<blocks((size_t)BATCH * S * 3), 256, 0, stream>>>(curC, fpsIdx, curN, S,
                                                                        coordsL[l + 1]);
    int Ctot = 2 * MLP[l][2];
    for (int sc = 0; sc < 2; ++sc) {
      int K = NS[l][sc];
      float r = RAD[l][sc];
      query_ball_k<<<blocks((size_t)BATCH * S * 32), 256, 0, stream>>>(
          curC, coordsL[l + 1], curN, S, K, r * r, gidx);
      int inC = INCH[l];
      int cinP0 = pad32i(inC + 3);
      int j0 = (l * 2 + sc) * 3;
      int M = S * K;
      int cout2 = MLP[l][2], coutP2 = pad32i(cout2);
      for (int b = 0; b < BATCH; ++b) {
        group_feat_k<<<blocks((size_t)M * cinP0), 256, 0, stream>>>(
            curC, coordsL[l + 1], curP, gidx, b, curN, S, K, inC, cinP0, FEAT);
        gemmLaunch(FEAT, j0 + 0, M, BUF1, nullptr, 1);
        gemmLaunch(BUF1, j0 + 1, M, BUF2, nullptr, 1);
        gemmLaunch(BUF2, j0 + 2, M, BUF1, nullptr, 1);
        maxpool_k<<<blocks((size_t)S * cout2), 256, 0, stream>>>(
            BUF1, S, K, cout2, coutP2, pooled[l + 1], b, Ctot, sc * cout2);
      }
    }
    curC = coordsL[l + 1];
    curP = pooled[l + 1];
    curN = S;
  }

  // ---- head MLP ----
  {
    int M = BATCH * 512;
    f32_to_bf16_k<<<blocks((size_t)M * 512), 256, 0, stream>>>(pooled[4], M, 512, 512, FEAT);
    const uint16_t* hin = FEAT;
    for (int i = 0; i < 5; ++i) {
      uint16_t* ob = (i & 1) ? BUF2 : BUF1;
      float* of = (i == 4) ? logits : nullptr;
      gemmLaunch(hin, 24 + i, M, ob, of, (i < 4) ? 1 : 0);
      hin = ob;
    }
  }

  // ---- softmax over points, skeleton einsum, transposed weights ----
  softmax_s_k<<<BATCH * 100, 128, 0, stream>>>(logits, wbuf, 512, 100);
  float* out = (float*)d_out;
  skel_out_k<<<blocks((size_t)BATCH * 100 * 3), 256, 0, stream>>>(wbuf, coordsL[4], out, 512, 100);
  w_transpose_k<<<blocks((size_t)BATCH * 100 * 512), 256, 0, stream>>>(
      wbuf, out + (size_t)BATCH * 100 * 3, 512, 100);
}
